// MultiHeadAttention_12567074308837
// MI455X (gfx1250) — matrix operand order repaired and checked by reading, not compiled, not run on hardware
//
#include <hip/hip_runtime.h>
#include <math.h>

// Problem constants (from reference): B=4, S=2048, D=1024, H=16, A=O=64
#define B_ 4
#define S_ 2048
#define D_ 1024
#define H_ 16
#define AO_ 64
#define ATTN_WAVES 4   // waves per attention block (share K/V tiles in LDS)

typedef __attribute__((ext_vector_type(16))) _Float16 v16h;
typedef __attribute__((ext_vector_type(8)))  _Float16 v8h;
typedef __attribute__((ext_vector_type(8)))  float    v8f;
typedef __attribute__((ext_vector_type(4)))  float    v4f;
typedef __attribute__((ext_vector_type(4)))  unsigned int v4u;
typedef __attribute__((ext_vector_type(8)))  int      v8i;
typedef __attribute__((ext_vector_type(4)))  int      v4i;

// ---------------------------------------------------------------------------
// WMMA fragment maps as used in this file (wave32); the A/B k order is this code's own choice:
//  A 16x32 f16 : lane&15 = M; k = hw + 16*hi (one contiguous 16-element run,
//                matching the B operand order)
//  B 32x16 f16 : lane&15 = N; k = hw + 16*hi  (one contiguous 16-element run
//                when source is [N x K] row-major)
//  C/D 16x16 f32: VGPR j -> M = j + 8*hi, N = lane&15
// ---------------------------------------------------------------------------
__device__ __forceinline__ int kB_idx(int lane, int hw) { return hw + (lane & 16); }

// A-fragment from row-major f16 [16 x >=32] tile: two v8h loads per lane
__device__ __forceinline__ v16h loadA_f16v(const _Float16* base, int ld, int lane) {
  const _Float16* p = base + (size_t)(lane & 15) * ld + (lane & 16);
  v8h lo = *(const v8h*)(p);
  v8h hi = *(const v8h*)(p + 8);
  v16h f;
#pragma unroll
  for (int i = 0; i < 8; ++i) { f[i] = lo[i]; f[8 + i] = hi[i]; }
  return f;
}
// A-fragment from row-major fp32 tile, inline convert (four v4f loads)
__device__ __forceinline__ v16h loadA_f32v(const float* base, int ld, int lane) {
  const float* p = base + (size_t)(lane & 15) * ld + (lane & 16);
  v4f a = *(const v4f*)(p);
  v4f b = *(const v4f*)(p + 4);
  v4f c = *(const v4f*)(p + 8);
  v4f d = *(const v4f*)(p + 12);
  v16h f;
#pragma unroll
  for (int i = 0; i < 4; ++i) {
    f[i] = (_Float16)a[i]; f[4 + i] = (_Float16)b[i];
    f[8 + i] = (_Float16)c[i]; f[12 + i] = (_Float16)d[i];
  }
  return f;
}
// B-fragment from row-major [N x K] f16 source (contiguous: two v8h loads)
__device__ __forceinline__ v16h loadB_NK_f16v(const _Float16* base, int ld, int lane) {
  const _Float16* p = base + (size_t)(lane & 15) * ld + (lane & 16);
  v8h lo = *(const v8h*)(p);
  v8h hi = *(const v8h*)(p + 8);
  v16h f;
#pragma unroll
  for (int i = 0; i < 8; ++i) { f[i] = lo[i]; f[8 + i] = hi[i]; }
  return f;
}
// B-fragment from row-major [K x N] f16 source (strided; used for V tile in LDS)
__device__ __forceinline__ v16h loadB_KN_f16(const _Float16* base, int ld, int lane) {
  v16h f; const int n = lane & 15;
#pragma unroll
  for (int i = 0; i < 16; ++i) f[i] = base[(size_t)kB_idx(lane, i) * ld + n];
  return f;
}

__device__ __forceinline__ v8f wmma_f16(v16h a, v16h b, v8f c) {
  return __builtin_amdgcn_wmma_f32_16x16x32_f16(false, a, false, b, (short)0, c,
                                                false, false);
}

// ---------------------------------------------------------------------------
// Tensor Data Mover: 1D tile load Global -> LDS (D# per ISA 08_async_tensor §8)
// 6-arg builtin on this toolchain:
//   (uint32x4 g0, int32x8 g1, int32x4 g2, int32x4 g3, int32x8 g4, i32 cpol)
// ---------------------------------------------------------------------------
#define HAVE_TDM __has_builtin(__builtin_amdgcn_tensor_load_to_lds)

#if HAVE_TDM
__device__ __forceinline__ void tdm_load_1d(unsigned lds_byte_off, const void* gsrc,
                                            unsigned tile_elems, unsigned tensor_elems) {
  const unsigned long long ga = (unsigned long long)(size_t)gsrc;
  v4u g0;
  g0.x = 1u;                                              // count=1, user D#
  g0.y = lds_byte_off;                                    // LDS byte address
  g0.z = (unsigned)ga;                                    // global addr [31:0]
  g0.w = (unsigned)((ga >> 32) & 0x01ffffffull) | (2u << 30);  // addr[56:32]|type=2
  v8i g1;
  g1[0] = 1 << 16;                                        // data_size=1 (2 bytes)
  g1[1] = (int)((tensor_elems & 0xffffu) << 16);          // tensor_dim0 lo16
  g1[2] = (int)((tensor_elems >> 16) & 0xffffu);          // tensor_dim0 hi16
  g1[3] = (int)((tile_elems & 0xffffu) << 16);            // tile_dim0 (1D tile)
  g1[4] = 0;                                              // tile_dim1/2 unused
  g1[5] = (int)tensor_elems;                              // dim0 stride lo32
  g1[6] = 0;
  g1[7] = 0;
  const v4i z4 = {0, 0, 0, 0};
  const v8i z8 = {0, 0, 0, 0, 0, 0, 0, 0};
  __builtin_amdgcn_tensor_load_to_lds(g0, g1, z4, z4, z8, 0);
}
#endif

__device__ __forceinline__ void wait_tensorcnt(int n) {
#if __has_builtin(__builtin_amdgcn_s_wait_tensorcnt)
  if (n == 0) __builtin_amdgcn_s_wait_tensorcnt(0);
  else        __builtin_amdgcn_s_wait_tensorcnt(2);
#else
  if (n == 0) asm volatile("s_wait_tensorcnt 0x0" ::: "memory");
  else        asm volatile("s_wait_tensorcnt 0x2" ::: "memory");
#endif
}

// ---------------------------------------------------------------------------
// Kernel 0: transpose+convert weights fp32 [H,K,N] -> f16 [H,N,K]
// ---------------------------------------------------------------------------
__global__ void wt_transpose_kernel(const float* __restrict__ W,
                                    _Float16* __restrict__ WT,
                                    int Dd, int Nn, int total) {
  const int idx = blockIdx.x * blockDim.x + threadIdx.x;
  if (idx >= total) return;
  const int n = idx % Nn;
  const int rem = idx / Nn;
  const int k = rem % Dd;
  const int h = rem / Dd;
  WT[((size_t)h * Nn + n) * Dd + k] = (_Float16)W[idx];
}

// ---------------------------------------------------------------------------
// Kernel 1: QKV projection. One wave per (16-row tile, head).
//   Y[b,h,s,:] = X[b,s,:] @ W[h] + bias[h].  WT is f16 [H, 64, D] (N-major).
// ---------------------------------------------------------------------------
__global__ void __launch_bounds__(32)
qkv_proj_kernel(const float* __restrict__ X, const _Float16* __restrict__ WT,
                const float* __restrict__ bias, _Float16* __restrict__ Y) {
  const int lane = threadIdx.x;
  const int tile = blockIdx.x;            // 16-row tile over B*S
  const int h    = blockIdx.y;
  const _Float16* WTh = WT + (size_t)h * AO_ * D_;
  const float* Xt = X + (size_t)tile * 16 * D_;

  v8f acc[4] = {};
  for (int kb = 0; kb < D_; kb += 32) {
    if (kb + 32 < D_)   // prefetch next X K-slab (global_prefetch_b8)
      __builtin_prefetch(Xt + (size_t)(lane & 15) * D_ + kb + 32, 0, 3);
    v16h a  = loadA_f32v(Xt + kb, D_, lane);
    v16h b0 = loadB_NK_f16v(WTh + (size_t)0 * 16 * D_ + kb, D_, lane);
    v16h b1 = loadB_NK_f16v(WTh + (size_t)1 * 16 * D_ + kb, D_, lane);
    v16h b2 = loadB_NK_f16v(WTh + (size_t)2 * 16 * D_ + kb, D_, lane);
    v16h b3 = loadB_NK_f16v(WTh + (size_t)3 * 16 * D_ + kb, D_, lane);
    acc[0] = wmma_f16(a, b0, acc[0]);
    acc[1] = wmma_f16(a, b1, acc[1]);
    acc[2] = wmma_f16(a, b2, acc[2]);
    acc[3] = wmma_f16(a, b3, acc[3]);
  }
  const int n_lo = lane & 15;
  const int hi   = (lane >> 4) << 3;
#pragma unroll
  for (int nb = 0; nb < 4; ++nb) {
    const int col = nb * 16 + n_lo;
    const float bv = bias[h * AO_ + col];
#pragma unroll
    for (int j = 0; j < 8; ++j) {
      const int row = tile * 16 + j + hi;              // flat (b*S + s)
      const int bb = row / S_, s = row % S_;
      Y[(((size_t)bb * H_ + h) * S_ + s) * AO_ + col] = (_Float16)(acc[nb][j] + bv);
    }
  }
}

// ---------------------------------------------------------------------------
// Kernel 2: flash attention. 4 waves/block; each wave owns one 16-query tile.
// K/V 32-key tiles are double-buffered in LDS: the Tensor Data Mover fetches
// tile i+1 while the block computes on tile i (s_wait_tensorcnt 2 = previous
// pair resident, since TDM ops complete in order per wave).
// Q/K/V f16 [B,H,S,64]; CTX f16 [B,S,H*64].
// ---------------------------------------------------------------------------
__global__ void __launch_bounds__(ATTN_WAVES * 32)
attn_kernel(const _Float16* __restrict__ Q, const _Float16* __restrict__ Kmat,
            const _Float16* __restrict__ V, _Float16* __restrict__ CTX) {
  __shared__ alignas(16) _Float16 sK[2][32 * AO_];              // 8 KB
  __shared__ alignas(16) _Float16 sV[2][32 * AO_];              // 8 KB
  __shared__ alignas(16) _Float16 sP[ATTN_WAVES * 16 * 32];     // 4 KB

  const int lane = threadIdx.x & 31;
  const int wave = threadIdx.x >> 5;
  const int qt = blockIdx.x * ATTN_WAVES + wave;
  const int h = blockIdx.y, b = blockIdx.z;
  const size_t bh = ((size_t)b * H_ + h) * S_;
  const _Float16* Qb = Q    + bh * AO_;
  const _Float16* Kb = Kmat + bh * AO_;
  const _Float16* Vb = V    + bh * AO_;
  _Float16* sPw = sP + wave * 16 * 32;

  // Q fragments for this wave's 16-row tile (A=64 -> two K=32 frags)
  const v16h q0 = loadA_f16v(Qb + (size_t)qt * 16 * AO_ + 0,  AO_, lane);
  const v16h q1 = loadA_f16v(Qb + (size_t)qt * 16 * AO_ + 32, AO_, lane);

  v8f acc[4] = {};
  float mrow[8], lrow[8];
#pragma unroll
  for (int j = 0; j < 8; ++j) { mrow[j] = -1e30f; lrow[j] = 0.0f; }
  const float scale = 0.125f;       // 1/sqrt(64)
  const int n_lo = lane & 15;
  const int hi   = (lane >> 4) << 3;

#if HAVE_TDM
  if (wave == 0) {   // prologue: stage tile 0 into buffer 0
    tdm_load_1d((unsigned)(size_t)(void*)sK[0], Kb, 32u * AO_, (unsigned)(S_ * AO_));
    tdm_load_1d((unsigned)(size_t)(void*)sV[0], Vb, 32u * AO_, (unsigned)(S_ * AO_));
  }
#endif

  for (int t = 0; t < S_; t += 32) {
    const int cur = (t >> 5) & 1;
#if HAVE_TDM
    if (wave == 0) {
      if (t + 32 < S_) {   // prefetch next tile into other buffer, then wait
        tdm_load_1d((unsigned)(size_t)(void*)sK[cur ^ 1],
                    Kb + (size_t)(t + 32) * AO_, 32u * AO_,
                    (unsigned)((S_ - t - 32) * AO_));
        tdm_load_1d((unsigned)(size_t)(void*)sV[cur ^ 1],
                    Vb + (size_t)(t + 32) * AO_, 32u * AO_,
                    (unsigned)((S_ - t - 32) * AO_));
        wait_tensorcnt(2);   // <=2 outstanding => current tile's pair resident
      } else {
        wait_tensorcnt(0);
      }
    }
#else
    {
      const v8h* gk = (const v8h*)(Kb + (size_t)t * AO_);
      const v8h* gv = (const v8h*)(Vb + (size_t)t * AO_);
      v8h* lk = (v8h*)sK[cur];
      v8h* lv = (v8h*)sV[cur];
      for (int i = threadIdx.x; i < (32 * AO_) / 8; i += ATTN_WAVES * 32) {
        lk[i] = gk[i];
        lv[i] = gv[i];
      }
    }
#endif
    __syncthreads();                 // current K/V tile visible to all waves
    const _Float16* kT = sK[cur];
    const _Float16* vT = sV[cur];

    // scores: S0 = keys [t, t+16), S1 = keys [t+16, t+32)   (kT is [N=t x K=a])
    v8f s0 = {}, s1 = {};
    s0 = wmma_f16(q0, loadB_NK_f16v(kT + 0,             AO_, lane), s0);
    s0 = wmma_f16(q1, loadB_NK_f16v(kT + 32,            AO_, lane), s0);
    s1 = wmma_f16(q0, loadB_NK_f16v(kT + 16 * AO_,      AO_, lane), s1);
    s1 = wmma_f16(q1, loadB_NK_f16v(kT + 16 * AO_ + 32, AO_, lane), s1);

    // online softmax (row m = j + 8*hi lives across the 16 lanes of a half)
#pragma unroll
    for (int j = 0; j < 8; ++j) {
      float x0 = s0[j] * scale;
      float x1 = s1[j] * scale;
      float mx = fmaxf(x0, x1);
#pragma unroll
      for (int off = 1; off < 16; off <<= 1) mx = fmaxf(mx, __shfl_xor(mx, off, 32));
      const float mnew = fmaxf(mrow[j], mx);
      const float corr = __expf(mrow[j] - mnew);
      const float p0 = __expf(x0 - mnew);
      const float p1 = __expf(x1 - mnew);
      float rs = p0 + p1;
#pragma unroll
      for (int off = 1; off < 16; off <<= 1) rs += __shfl_xor(rs, off, 32);
      lrow[j] = lrow[j] * corr + rs;
      mrow[j] = mnew;
      acc[0][j] *= corr; acc[1][j] *= corr; acc[2][j] *= corr; acc[3][j] *= corr;
      const int m = j + hi;
      sPw[m * 32 + n_lo]      = (_Float16)p0;   // wave-private; DS in-order
      sPw[m * 32 + 16 + n_lo] = (_Float16)p1;
    }

    // ctx += P[16x32] x V[t:t+32, 0:64]
    const v16h p = loadA_f16v(sPw, 32, lane);
#pragma unroll
    for (int nb = 0; nb < 4; ++nb) {
      v16h vb = loadB_KN_f16(vT + nb * 16, AO_, lane);
      acc[nb] = wmma_f16(p, vb, acc[nb]);
    }
    __syncthreads();                 // all waves done with buf[cur] before it is
                                     // refilled at the next iteration's top
  }

  // finalize: divide by softmax denominator, store concat-head layout
#pragma unroll
  for (int nb = 0; nb < 4; ++nb) {
#pragma unroll
    for (int j = 0; j < 8; ++j) {
      const int s = qt * 16 + j + hi;
      CTX[((size_t)b * S_ + s) * (H_ * AO_) + h * AO_ + nb * 16 + n_lo] =
          (_Float16)(acc[nb][j] / lrow[j]);
    }
  }
}

// ---------------------------------------------------------------------------
// Kernel 3: output projection. y[B*S,64] = CTX[B*S,1024] @ Wo + bo (fp32 out)
// WoT is f16 [64, 1024] (N-major).
// ---------------------------------------------------------------------------
__global__ void __launch_bounds__(32)
out_proj_kernel(const _Float16* __restrict__ X, const _Float16* __restrict__ WoT,
                const float* __restrict__ bo, float* __restrict__ Y) {
  const int lane = threadIdx.x;
  const int tile = blockIdx.x;
  const int KD = H_ * AO_;  // 1024
  const _Float16* Xt = X + (size_t)tile * 16 * KD;

  v8f acc[4] = {};
  for (int kb = 0; kb < KD; kb += 32) {
    v16h a  = loadA_f16v(Xt + kb, KD, lane);
    v16h b0 = loadB_NK_f16v(WoT + (size_t)0 * 16 * KD + kb, KD, lane);
    v16h b1 = loadB_NK_f16v(WoT + (size_t)1 * 16 * KD + kb, KD, lane);
    v16h b2 = loadB_NK_f16v(WoT + (size_t)2 * 16 * KD + kb, KD, lane);
    v16h b3 = loadB_NK_f16v(WoT + (size_t)3 * 16 * KD + kb, KD, lane);
    acc[0] = wmma_f16(a, b0, acc[0]);
    acc[1] = wmma_f16(a, b1, acc[1]);
    acc[2] = wmma_f16(a, b2, acc[2]);
    acc[3] = wmma_f16(a, b3, acc[3]);
  }
  const int n_lo = lane & 15;
  const int hi   = (lane >> 4) << 3;
#pragma unroll
  for (int nb = 0; nb < 4; ++nb) {
    const int col = nb * 16 + n_lo;
    const float bv = bo[col];
#pragma unroll
    for (int j = 0; j < 8; ++j) {
      const int row = tile * 16 + j + hi;
      Y[(size_t)row * AO_ + col] = acc[nb][j] + bv;
    }
  }
}

// ---------------------------------------------------------------------------
// Launcher. Workspace (bytes):
//   Q16 | K16 | V16 | CTX16  (16 MiB each)  then WqT/WkT/WvT (2 MiB) | WoT
// ---------------------------------------------------------------------------
extern "C" void kernel_launch(void* const* d_in, const int* in_sizes, int n_in,
                              void* d_out, int out_size, void* d_ws, size_t ws_size,
                              hipStream_t stream) {
  (void)in_sizes; (void)n_in; (void)out_size; (void)ws_size;

  const float* cur = (const float*)d_in[0];
  const float* ctx = (const float*)d_in[1];
  const float* Wq  = (const float*)d_in[2];
  const float* bq  = (const float*)d_in[3];
  const float* Wk  = (const float*)d_in[4];
  const float* bk  = (const float*)d_in[5];
  const float* Wv  = (const float*)d_in[6];
  const float* bv  = (const float*)d_in[7];
  const float* Wo  = (const float*)d_in[8];
  const float* bo  = (const float*)d_in[9];
  float* out = (float*)d_out;

  const size_t qkv_bytes = (size_t)B_ * H_ * S_ * AO_ * 2;   // 16 MiB
  const size_t wqkv_bytes = (size_t)H_ * D_ * AO_ * 2;       // 2 MiB
  char* ws = (char*)d_ws;
  _Float16* Q16   = (_Float16*)(ws);
  _Float16* K16   = (_Float16*)(ws + qkv_bytes);
  _Float16* V16   = (_Float16*)(ws + 2 * qkv_bytes);
  _Float16* CTX16 = (_Float16*)(ws + 3 * qkv_bytes);
  _Float16* WqT   = (_Float16*)(ws + 4 * qkv_bytes);
  _Float16* WkT   = (_Float16*)(ws + 4 * qkv_bytes + wqkv_bytes);
  _Float16* WvT   = (_Float16*)(ws + 4 * qkv_bytes + 2 * wqkv_bytes);
  _Float16* WoT   = (_Float16*)(ws + 4 * qkv_bytes + 3 * wqkv_bytes);

  // 0) weight transpose+convert: [H,K,N] fp32 -> [H,N,K] f16
  const int wqkv_total = H_ * D_ * AO_;                      // 1,048,576
  const int wo_total = (H_ * AO_) * AO_;                     // 65,536
  wt_transpose_kernel<<<(wqkv_total + 255) / 256, 256, 0, stream>>>(Wq, WqT, D_, AO_, wqkv_total);
  wt_transpose_kernel<<<(wqkv_total + 255) / 256, 256, 0, stream>>>(Wk, WkT, D_, AO_, wqkv_total);
  wt_transpose_kernel<<<(wqkv_total + 255) / 256, 256, 0, stream>>>(Wv, WvT, D_, AO_, wqkv_total);
  wt_transpose_kernel<<<(wo_total + 255) / 256, 256, 0, stream>>>(Wo, WoT, H_ * AO_, AO_, wo_total);

  const int row_tiles = (B_ * S_) / 16;                      // 512

  // 1) QKV projections
  qkv_proj_kernel<<<dim3(row_tiles, H_), 32, 0, stream>>>(cur, WqT, bq, Q16);
  qkv_proj_kernel<<<dim3(row_tiles, H_), 32, 0, stream>>>(ctx, WkT, bk, K16);
  qkv_proj_kernel<<<dim3(row_tiles, H_), 32, 0, stream>>>(ctx, WvT, bv, V16);

  // 2) flash attention (4 waves/block, double-buffered TDM-staged K/V tiles)
  attn_kernel<<<dim3(S_ / 16 / ATTN_WAVES, H_, B_), ATTN_WAVES * 32, 0, stream>>>(
      Q16, K16, V16, CTX16);

  // 3) output projection
  out_proj_kernel<<<dim3(row_tiles), 32, 0, stream>>>(CTX16, WoT, bo, out);
}
